// Block_36155034698246
// MI455X (gfx1250) — matrix operand order repaired and checked by reading, not compiled, not run on hardware
//
#include <hip/hip_runtime.h>
#include <hip/hip_bf16.h>
#include <cstdint>
#include <cstddef>

// ---------------- fixed problem shape ----------------
#define Bn   4
#define Tn   2048
#define Cn   1024
#define Hn   4
#define HDn  256
#define DFFn 4096
#define BTn  (Bn * Tn)            // 8192 rows

typedef __bf16 bf16_t;
typedef __bf16 v16bf __attribute__((ext_vector_type(16)));
typedef __bf16 v8bf  __attribute__((ext_vector_type(8)));
typedef float  v8f   __attribute__((ext_vector_type(8)));
typedef int    i128_t __attribute__((vector_size(16)));   // 4 x i32

// ---------------- CDNA5 async LDS copy path ----------------
#if __has_builtin(__builtin_amdgcn_global_load_async_to_lds_b128)
#define ASYNC_LDS_OK 1
#endif

// copy 16B global -> LDS (async on CDNA5; sync fallback otherwise)
__device__ __forceinline__ void async_cp16(bf16_t* lds_dst, const bf16_t* g_src) {
#ifdef ASYNC_LDS_OK
  __builtin_amdgcn_global_load_async_to_lds_b128(
      (__attribute__((address_space(1))) i128_t*)(const_cast<bf16_t*>(g_src)),
      (__attribute__((address_space(3))) i128_t*)(lds_dst),
      /*offset=*/0, /*cpol=*/0);
#else
  *(v8bf*)lds_dst = *(const v8bf*)g_src;
#endif
}

template <int N>
__device__ __forceinline__ void wait_async() {
#ifdef ASYNC_LDS_OK
#if __has_builtin(__builtin_amdgcn_s_wait_asynccnt)
  __builtin_amdgcn_s_wait_asynccnt(N);
#else
  asm volatile("s_wait_asynccnt %0" ::"i"(N) : "memory");
#endif
#endif
}

// ---------------- WMMA helpers ----------------
__device__ __forceinline__ v8f wmma_bf16(v16bf a, v16bf b, v8f c) {
  return __builtin_amdgcn_wmma_f32_16x16x32_bf16(
      false, a, false, b, (short)0, c, false, false);
}

// A-fragment (ISA 16-bit A 16x32): lane row m = lane&15; two contiguous 8-elem
// chunks at k = half*8 and k = 16 + half*8
__device__ __forceinline__ v16bf ld_fragA(const bf16_t* rowp, int half) {
  union { v16bf v; v8bf h[2]; } u;
  u.h[0] = *(const v8bf*)(rowp + half * 8);
  u.h[1] = *(const v8bf*)(rowp + 16 + half * 8);
  return u.v;
}

// B-fragment from a transposed tile BT[n][k]: lane col n = lane&15;
// two contiguous 8-elem chunks at k = half*8 and k = 16 + half*8
__device__ __forceinline__ v16bf ld_fragB(const bf16_t* colp, int half) {
  union { v16bf v; v8bf h[2]; } u;
  u.h[0] = *(const v8bf*)(colp + half * 8);
  u.h[1] = *(const v8bf*)(colp + 16 + half * 8);
  return u.v;
}

__device__ __forceinline__ v8f vzero8() {
  v8f z = {0.f, 0.f, 0.f, 0.f, 0.f, 0.f, 0.f, 0.f};
  return z;
}

// ---------------- repack kernels (one-time, weights -> bf16 col-major) -------
// wq/wk/wv: [H][C][HD] fp32 -> [H*HD][C] bf16  (i.e. [N][K])
__global__ void repack_head_w(const float* __restrict__ in, bf16_t* __restrict__ out) {
  int i = blockIdx.x * 256 + threadIdx.x;       // H*C*HD = 1048576
  int h = i >> 18;                              // C*HD = 2^18
  int c = (i >> 8) & (Cn - 1);
  int d = i & (HDn - 1);
  out[(size_t)((h << 8) + d) * Cn + c] = (bf16_t)in[i];
}

// [K][N] fp32 -> [N][K] bf16
__global__ void transpose_w(const float* __restrict__ in, bf16_t* __restrict__ out,
                            int K, int N) {
  int i = blockIdx.x * 256 + threadIdx.x;
  if (i >= K * N) return;
  int k = i / N, n = i - k * N;                 // coalesced read
  out[(size_t)n * K + k] = (bf16_t)in[i];
}

// ---------------- layernorm (row of C=1024, 256 threads) ----------------
__global__ __launch_bounds__(256) void ln_kernel(const float* __restrict__ x,
                                                 const float* __restrict__ w,
                                                 const float* __restrict__ b,
                                                 bf16_t* __restrict__ out) {
  __shared__ float red[256];
  const int row = blockIdx.x, t = threadIdx.x;
  const float* xr = x + (size_t)row * Cn;
  float v[4], s = 0.f;
#pragma unroll
  for (int j = 0; j < 4; ++j) { v[j] = xr[t + j * 256]; s += v[j]; }
  red[t] = s; __syncthreads();
  for (int off = 128; off > 0; off >>= 1) {
    if (t < off) red[t] += red[t + off];
    __syncthreads();
  }
  const float mu = red[0] * (1.f / Cn);
  __syncthreads();
  float s2 = 0.f;
#pragma unroll
  for (int j = 0; j < 4; ++j) { float d = v[j] - mu; s2 += d * d; }
  red[t] = s2; __syncthreads();
  for (int off = 128; off > 0; off >>= 1) {
    if (t < off) red[t] += red[t + off];
    __syncthreads();
  }
  const float rstd = rsqrtf(red[0] * (1.f / Cn) + 1e-5f);
  bf16_t* orow = out + (size_t)row * Cn;
#pragma unroll
  for (int j = 0; j < 4; ++j) {
    int col = t + j * 256;
    orow[col] = (bf16_t)((v[j] - mu) * rstd * w[col] + b[col]);
  }
}

// ---------------- bf16 WMMA GEMM ----------------
// out = act(A[MxK] @ Wt[NxK]^T + bias) (+resid); block 128x128, 8 waves 4m x 2n
// Both A and Wt tiles are linear row copies -> async LDS, double buffered.
#define LDA_S 40                      // LDS row stride (bf16), mult of 8
#define GTILE (128 * LDA_S)

__device__ __forceinline__ void gemm_stage(bf16_t* Ab, bf16_t* Wb,
                                           const bf16_t* A, const bf16_t* Wt,
                                           int K, int k0, int m0, int n0, int t) {
  const int row = t >> 1, hf = t & 1;
  const bf16_t* ag = A  + (size_t)(m0 + row) * K + k0 + hf * 16;
  const bf16_t* wg = Wt + (size_t)(n0 + row) * K + k0 + hf * 16;
  bf16_t* al = Ab + row * LDA_S + hf * 16;
  bf16_t* wl = Wb + row * LDA_S + hf * 16;
  async_cp16(al,     ag);
  async_cp16(al + 8, ag + 8);
  async_cp16(wl,     wg);
  async_cp16(wl + 8, wg + 8);
}

template <bool HAS_BIAS, bool RELU, bool HAS_RES, bool OUT_F, bool OUT_B, bool OUT_BT>
__global__ __launch_bounds__(256) void gemm_bf16(
    const bf16_t* __restrict__ A, const bf16_t* __restrict__ Wt,
    int M, int N, int K,
    const float* __restrict__ bias, const float* __restrict__ resid,
    float* __restrict__ outf, bf16_t* __restrict__ outb, bf16_t* __restrict__ outbT) {
  __shared__ __align__(16) bf16_t As[2 * GTILE];
  __shared__ __align__(16) bf16_t Ws[2 * GTILE];
  const int t = threadIdx.x;
  const int lane = t & 31, wv = t >> 5;
  const int half = lane >> 4, l16 = lane & 15;
  const int wm = wv >> 1, wn = wv & 1;
  const int m0 = blockIdx.y * 128, n0 = blockIdx.x * 128;

  v8f acc[2][4];
#pragma unroll
  for (int i = 0; i < 2; ++i)
#pragma unroll
    for (int j = 0; j < 4; ++j) acc[i][j] = vzero8();

  const int nk = K >> 5;
  gemm_stage(As, Ws, A, Wt, K, 0, m0, n0, t);
  for (int kt = 0; kt < nk; ++kt) {
    bf16_t* Ab = As + (kt & 1) * GTILE;
    bf16_t* Wb = Ws + (kt & 1) * GTILE;
    if (kt + 1 < nk) {
      gemm_stage(As + ((kt + 1) & 1) * GTILE, Ws + ((kt + 1) & 1) * GTILE,
                 A, Wt, K, (kt + 1) * 32, m0, n0, t);
      wait_async<4>();
    } else {
      wait_async<0>();
    }
    __syncthreads();

    v16bf af[2], bfm[4];
#pragma unroll
    for (int i = 0; i < 2; ++i)
      af[i] = ld_fragA(Ab + (wm * 32 + i * 16 + l16) * LDA_S, half);
#pragma unroll
    for (int j = 0; j < 4; ++j)
      bfm[j] = ld_fragB(Wb + (wn * 64 + j * 16 + l16) * LDA_S, half);
#pragma unroll
    for (int i = 0; i < 2; ++i)
#pragma unroll
      for (int j = 0; j < 4; ++j)
        acc[i][j] = wmma_bf16(af[i], bfm[j], acc[i][j]);
    __syncthreads();
  }

  // epilogue: C/D layout row = g + 8*half, col = lane&15
#pragma unroll
  for (int j = 0; j < 4; ++j) {
    const int col = n0 + wn * 64 + j * 16 + l16;
    float bcol = 0.f;
    if (HAS_BIAS) bcol = bias[col];
#pragma unroll
    for (int i = 0; i < 2; ++i) {
#pragma unroll
      for (int g = 0; g < 8; ++g) {
        const int row = m0 + wm * 32 + i * 16 + g + 8 * half;
        float val = acc[i][j][g] + bcol;
        if (RELU) val = fmaxf(val, 0.f);
        const size_t idx = (size_t)row * N + col;
        if (HAS_RES) val += resid[idx];
        if (OUT_F) outf[idx] = val;
        if (OUT_B) outb[idx] = (bf16_t)val;
        if (OUT_BT) outbT[(size_t)col * M + row] = (bf16_t)val;
      }
    }
  }
}

// ---------------- flash attention (causal), bf16 WMMA ----------------
// grid: (T/128, B*H); 8 waves; wave owns 16 q-rows. V comes pre-transposed.
#define KS_S 264                       // 256 + 8 pad
#define VT_S 40                        // 32 + 8 pad
#define PS_S 40
#define KTILE (32 * KS_S)
#define VTILE (256 * VT_S)

__device__ __forceinline__ void attn_stage(bf16_t* Kb, bf16_t* Vb,
                                           const bf16_t* kg, const bf16_t* vTg,
                                           size_t kbase, size_t vbase, int s0, int t) {
  // K tile [s][d]: 32 x 256, linear rows
#pragma unroll
  for (int i = 0; i < 4; ++i) {
    int c = t + i * 256;
    int s = c >> 5, c8 = (c & 31) * 8;
    async_cp16(Kb + s * KS_S + c8, kg + kbase + (size_t)(s0 + s) * Cn + c8);
  }
  // V^T tile [d][s]: 256 x 32, linear rows from pre-transposed V [C][BT]
#pragma unroll
  for (int i = 0; i < 4; ++i) {
    int c = t + i * 256;
    int d = c >> 2, sc = (c & 3) * 8;
    async_cp16(Vb + d * VT_S + sc, vTg + vbase + (size_t)d * BTn + s0 + sc);
  }
}

__global__ __launch_bounds__(256) void attn_kernel(
    const bf16_t* __restrict__ qg, const bf16_t* __restrict__ kg,
    const bf16_t* __restrict__ vTg, bf16_t* __restrict__ og) {
  __shared__ __align__(16) bf16_t Ks[2 * KTILE];
  __shared__ __align__(16) bf16_t VT[2 * VTILE];
  __shared__ __align__(16) bf16_t Ps[128 * PS_S];
  const int t = threadIdx.x;
  const int lane = t & 31, w = t >> 5;
  const int half = lane >> 4, l16 = lane & 15;
  const int q0 = blockIdx.x * 128;
  const int bh = blockIdx.y;
  const int b = bh >> 2, h = bh & 3;
  const size_t base  = (size_t)b * Tn * Cn + (size_t)h * HDn;       // q/k/out
  const size_t vbase = (size_t)h * HDn * BTn + (size_t)b * Tn;      // V^T

  // Q fragments for this wave's 16 rows, in registers (8 d-steps)
  v16bf qf[8];
  {
    const bf16_t* qrow = qg + base + (size_t)(q0 + w * 16 + l16) * Cn;
#pragma unroll
    for (int dk = 0; dk < 8; ++dk) qf[dk] = ld_fragA(qrow + dk * 32, half);
  }

  v8f o[16];
#pragma unroll
  for (int dt = 0; dt < 16; ++dt) o[dt] = vzero8();
  float rowmax[8], rowsum[8];
#pragma unroll
  for (int g = 0; g < 8; ++g) { rowmax[g] = -1e30f; rowsum[g] = 0.f; }
  const float scale = 0.03125f;  // C^-0.5 = 1/32

  const int nsteps = (q0 + 128) >> 5;
  attn_stage(Ks, VT, kg, vTg, base, vbase, 0, t);
  for (int st = 0; st < nsteps; ++st) {
    const int s0 = st * 32;
    bf16_t* Kb = Ks + (st & 1) * KTILE;
    bf16_t* Vb = VT + (st & 1) * VTILE;
    if (st + 1 < nsteps) {
      attn_stage(Ks + ((st + 1) & 1) * KTILE, VT + ((st + 1) & 1) * VTILE,
                 kg, vTg, base, vbase, s0 + 32, t);
      wait_async<8>();
    } else {
      wait_async<0>();
    }
    __syncthreads();

    // ---- scores S[16 x 32]: 8 d-steps, 2 n-tiles
    v8f sacc[2];
    sacc[0] = vzero8(); sacc[1] = vzero8();
#pragma unroll
    for (int dk = 0; dk < 8; ++dk) {
#pragma unroll
      for (int j = 0; j < 2; ++j) {
        v16bf bk = ld_fragB(Kb + (j * 16 + l16) * KS_S + dk * 32, half);
        sacc[j] = wmma_bf16(qf[dk], bk, sacc[j]);
      }
    }

    // ---- online softmax (rows = g + 8*half, cols = l16 per j-tile)
    float corr[8];
#pragma unroll
    for (int g = 0; g < 8; ++g) {
      const int qrow = q0 + w * 16 + g + 8 * half;
      float sv0 = sacc[0][g] * scale;
      float sv1 = sacc[1][g] * scale;
      if (s0 + l16 > qrow)      sv0 = -1e30f;
      if (s0 + 16 + l16 > qrow) sv1 = -1e30f;
      float tm = fmaxf(sv0, sv1);
#pragma unroll
      for (int m = 1; m < 16; m <<= 1) tm = fmaxf(tm, __shfl_xor(tm, m, 32));
      const float mnew = fmaxf(rowmax[g], tm);
      const float c = __expf(rowmax[g] - mnew);
      rowmax[g] = mnew;
      corr[g] = c;
      const float p0 = __expf(sv0 - mnew);
      const float p1 = __expf(sv1 - mnew);
      float ps = p0 + p1;
#pragma unroll
      for (int m = 1; m < 16; m <<= 1) ps += __shfl_xor(ps, m, 32);
      rowsum[g] = rowsum[g] * c + ps;
      Ps[(w * 16 + g + 8 * half) * PS_S + l16]      = (bf16_t)p0;
      Ps[(w * 16 + g + 8 * half) * PS_S + 16 + l16] = (bf16_t)p1;
    }
#pragma unroll
    for (int dt = 0; dt < 16; ++dt)
#pragma unroll
      for (int g = 0; g < 8; ++g) o[dt][g] *= corr[g];

    // ---- O += P(16x32) . V(32x256): 16 d-tiles
    v16bf ap = ld_fragA(Ps + (w * 16 + l16) * PS_S, half);
#pragma unroll
    for (int dt = 0; dt < 16; ++dt) {
      v16bf bv = ld_fragB(Vb + (dt * 16 + l16) * VT_S, half);
      o[dt] = wmma_bf16(ap, bv, o[dt]);
    }
    __syncthreads();
  }

  // write normalized output, layout [b, t, h*HD + d]
#pragma unroll
  for (int dt = 0; dt < 16; ++dt) {
#pragma unroll
    for (int g = 0; g < 8; ++g) {
      const int row = q0 + w * 16 + g + 8 * half;
      og[base + (size_t)row * Cn + dt * 16 + l16] = (bf16_t)(o[dt][g] / rowsum[g]);
    }
  }
}

// ---------------- host orchestration ----------------
extern "C" void kernel_launch(void* const* d_in, const int* in_sizes, int n_in,
                              void* d_out, int out_size, void* d_ws, size_t ws_size,
                              hipStream_t stream) {
  (void)in_sizes; (void)n_in; (void)out_size; (void)ws_size;
  const float* x      = (const float*)d_in[0];
  const float* ln1_w  = (const float*)d_in[1];
  const float* ln1_b  = (const float*)d_in[2];
  const float* wq     = (const float*)d_in[3];
  const float* wk     = (const float*)d_in[4];
  const float* wv     = (const float*)d_in[5];
  const float* proj_w = (const float*)d_in[6];
  const float* proj_b = (const float*)d_in[7];
  const float* ln2_w  = (const float*)d_in[8];
  const float* ln2_b  = (const float*)d_in[9];
  const float* w1     = (const float*)d_in[10];
  const float* b1     = (const float*)d_in[11];
  const float* w2     = (const float*)d_in[12];
  const float* b2     = (const float*)d_in[13];

  char* ws = (char*)d_ws;
  constexpr size_t MBv = 1u << 20;
  bf16_t* h1  = (bf16_t*)(ws + 0 * MBv);       // 16MB
  bf16_t* qb  = (bf16_t*)(ws + 16 * MBv);      // 16MB
  bf16_t* kb  = (bf16_t*)(ws + 32 * MBv);      // 16MB
  bf16_t* vTb = (bf16_t*)(ws + 48 * MBv);      // 16MB, layout [C][B*T]
  bf16_t* ff1 = (bf16_t*)(ws + 0 * MBv);       // 64MB, aliases h1/q/k/v (dead)
  bf16_t* ao  = (bf16_t*)(ws + 64 * MBv);      // 16MB
  float*  xm  = (float*) (ws + 80 * MBv);      // 32MB
  bf16_t* h2  = (bf16_t*)(ws + 112 * MBv);     // 16MB
  bf16_t* wqb = (bf16_t*)(ws + 128 * MBv);     // 2MB each, col-major [N][K]
  bf16_t* wkb = (bf16_t*)(ws + 130 * MBv);
  bf16_t* wvb = (bf16_t*)(ws + 132 * MBv);
  bf16_t* pwb = (bf16_t*)(ws + 134 * MBv);
  bf16_t* w1b = (bf16_t*)(ws + 136 * MBv);     // 8MB
  bf16_t* w2b = (bf16_t*)(ws + 144 * MBv);     // 8MB -> 152MB total
  float* out  = (float*)d_out;

  // one-time weight repacks (all -> bf16 col-major [N][K])
  repack_head_w<<<(Hn * Cn * HDn) / 256, 256, 0, stream>>>(wq, wqb);
  repack_head_w<<<(Hn * Cn * HDn) / 256, 256, 0, stream>>>(wk, wkb);
  repack_head_w<<<(Hn * Cn * HDn) / 256, 256, 0, stream>>>(wv, wvb);
  transpose_w<<<(Cn * Cn) / 256, 256, 0, stream>>>(proj_w, pwb, Cn, Cn);
  transpose_w<<<(Cn * DFFn) / 256, 256, 0, stream>>>(w1, w1b, Cn, DFFn);
  transpose_w<<<(DFFn * Cn) / 256, 256, 0, stream>>>(w2, w2b, DFFn, Cn);

  // LN1
  ln_kernel<<<BTn, 256, 0, stream>>>(x, ln1_w, ln1_b, h1);
  // QKV projections (V written pre-transposed for attention)
  dim3 gqkv(Cn / 128, BTn / 128);
  gemm_bf16<false, false, false, false, true, false><<<gqkv, 256, 0, stream>>>(
      h1, wqb, BTn, Cn, Cn, nullptr, nullptr, nullptr, qb, nullptr);
  gemm_bf16<false, false, false, false, true, false><<<gqkv, 256, 0, stream>>>(
      h1, wkb, BTn, Cn, Cn, nullptr, nullptr, nullptr, kb, nullptr);
  gemm_bf16<false, false, false, false, false, true><<<gqkv, 256, 0, stream>>>(
      h1, wvb, BTn, Cn, Cn, nullptr, nullptr, nullptr, nullptr, vTb);
  // attention
  attn_kernel<<<dim3(Tn / 128, Bn * Hn), 256, 0, stream>>>(qb, kb, vTb, ao);
  // output projection + bias + residual(x) -> xm (fp32)
  gemm_bf16<true, false, true, true, false, false><<<gqkv, 256, 0, stream>>>(
      ao, pwb, BTn, Cn, Cn, proj_b, x, xm, nullptr, nullptr);
  // LN2
  ln_kernel<<<BTn, 256, 0, stream>>>(xm, ln2_w, ln2_b, h2);
  // FFN
  gemm_bf16<true, true, false, false, true, false><<<dim3(DFFn / 128, BTn / 128), 256, 0, stream>>>(
      h2, w1b, BTn, DFFn, Cn, b1, nullptr, nullptr, ff1, nullptr);
  gemm_bf16<true, false, true, true, false, false><<<dim3(Cn / 128, BTn / 128), 256, 0, stream>>>(
      ff1, w2b, BTn, Cn, DFFn, b2, xm, out, nullptr, nullptr);
}
